// STRNN_41815801594609
// MI455X (gfx1250) — matrix operand order repaired and checked by reading, not compiled, not run on hardware
//
#include <hip/hip_runtime.h>
#include <hip/hip_bf16.h>

typedef __attribute__((ext_vector_type(16))) _Float16 v16h;
typedef __attribute__((ext_vector_type(8)))  float    v8f;

#define HDIM 64
#define TMAX 20

// Hardware tanh (CDNA5 V_TANH_F32 transcendental) instead of branchy OCML tanhf.
__device__ __forceinline__ float hw_tanhf(float v) {
#if __has_builtin(__builtin_amdgcn_tanhf)
  return __builtin_amdgcn_tanhf(v);
#else
  float r;
  // TRANS op: one independent op (v_nop) before the result may be consumed.
  asm volatile("v_tanh_f32 %0, %1\n\tv_nop" : "=v"(r) : "v"(v));
  return r;
#endif
}

// Same-wave LDS ordering point: LDS ops from one wave complete in order, so a
// dscnt wait + compiler fence is sufficient for intra-wave lane data exchange
// (each wave owns a private LDS region -> no workgroup barrier needed).
__device__ __forceinline__ void lds_fence() {
  asm volatile("s_wait_dscnt 0x0" ::: "memory");
}

// ---------------------------------------------------------------------------
// K0: fold GCN+FC:  u[k] = sum_h W_fc[h] * W_gcn[h,k];  cbias = b_gcn.W_fc + b_fc
// ---------------------------------------------------------------------------
__global__ void strnn_prep(const float* __restrict__ Wgcn, const float* __restrict__ bgcn,
                           const float* __restrict__ Wfc,  const float* __restrict__ bfc,
                           float* __restrict__ ws) {
  int n = threadIdx.x;
  if (n < HDIM) {
    float u = 0.0f;
    for (int h = 0; h < HDIM; ++h) u = fmaf(Wfc[h], Wgcn[h * HDIM + n], u);
    ws[n] = u;
  }
  if (n == 0) {
    float cb = bfc[0];
    for (int h = 0; h < HDIM; ++h) cb = fmaf(bgcn[h], Wfc[h], cb);
    ws[64] = cb;
  }
}

// ---------------------------------------------------------------------------
// K1: RNN over T steps via v_wmma_f32_16x16x32_f16; one wave = 16 nodes.
//     h (16x64) kept as two 16x32 f16 A-fragments; W_hh^T held as eight
//     32x16 B-fragments in registers. D->A relayout via per-wave LDS tile.
//     Fuses final s[node] = h_last . u  (collapsed GCN+FC vector).
// ---------------------------------------------------------------------------
__global__ __launch_bounds__(256)
void strnn_rnn(const float* __restrict__ x,   const float* __restrict__ Wih,
               const float* __restrict__ bih, const float* __restrict__ Whh,
               const float* __restrict__ bhh, const float* __restrict__ ws,
               float* __restrict__ s_out, int N, int T)
{
  __shared__ _Float16 h_lds[8][16 * HDIM];   // 16 KB: per-wave h staging
  __shared__ float    x_lds[8][16 * TMAX];   // 10 KB: per-wave x tile

  const int lane = threadIdx.x & 31;
  const int wv   = threadIdx.x >> 5;
  const int base = (blockIdx.x * 8 + wv) * 16;   // first node of this wave's tile
  const int n    = lane & 15;                    // lane's column within a 16-wide tile
  const int hi   = lane >> 4;                    // half-wave selector

  // Stage x tile (16 nodes x T scalars); pad out-of-range nodes with 0.
  for (int i = lane; i < 16 * T; i += 32) {
    int m = i / T, t = i - m * T;
    int node = base + m;
    x_lds[wv][m * T + t] = (node < N) ? x[node * T + t] : 0.0f;
  }

  // B fragments of W_hh^T:  B[k][n] = Whh[(16j+n)*64 + 32c + 8*hi + e + (e&8)]
  // (two runs of 8 contiguous floats -> 32B reads, f32 -> f16 convert)
  v16h B[4][2];
#pragma unroll
  for (int j = 0; j < 4; ++j) {
#pragma unroll
    for (int c = 0; c < 2; ++c) {
      const float* p = Whh + (16 * j + n) * HDIM + 32 * c + 8 * hi;
      v16h b;
#pragma unroll
      for (int e = 0; e < 16; ++e) b[e] = (_Float16)p[e + (e & 8)];
      B[j][c] = b;
    }
  }

  // Per-lane input weight and fused bias for each of the 4 output column tiles.
  float wihv[4], biasv[4];
#pragma unroll
  for (int j = 0; j < 4; ++j) {
    int col = 16 * j + n;
    wihv[j]  = Wih[col];
    biasv[j] = bih[col] + bhh[col];
  }

  v16h A0 = {}, A1 = {};   // h = 0 initial state, A-fragment layout

  lds_fence();             // x_lds staging visible to this wave

  for (int t = 0; t < T; ++t) {
#pragma unroll
    for (int j = 0; j < 4; ++j) {
      v8f acc = {};
      // D = A(16x32) x B(32x16) + C, chained over K = 0..63
      acc = __builtin_amdgcn_wmma_f32_16x16x32_f16(false, A0, false, B[j][0],
                                                   (short)0, acc, false, false);
      acc = __builtin_amdgcn_wmma_f32_16x16x32_f16(false, A1, false, B[j][1],
                                                   (short)0, acc, false, false);
#pragma unroll
      for (int r = 0; r < 8; ++r) {
        int m = r + 8 * hi;  // D layout: VGPR r -> row r (+8 for upper half-wave)
        float val = fmaf(x_lds[wv][m * T + t], wihv[j], acc[r] + biasv[j]);
        h_lds[wv][m * HDIM + 16 * j + n] = (_Float16)hw_tanhf(val);
      }
    }
    lds_fence();           // h stores of this wave complete, in-order w.r.t. loads
    // Reload h as A fragments: lane holds row m=n, k = 16*(e>>3) + 8*hi + (e&7) (+32c)
    {
      const int m = n;
      v16h a0, a1;
#pragma unroll
      for (int e = 0; e < 8; ++e) {
        a0[e]     = h_lds[wv][m * HDIM +  0 + 8 * hi + e];
        a0[e + 8] = h_lds[wv][m * HDIM + 16 + 8 * hi + e];
        a1[e]     = h_lds[wv][m * HDIM + 32 + 8 * hi + e];
        a1[e + 8] = h_lds[wv][m * HDIM + 48 + 8 * hi + e];
      }
      A0 = a0; A1 = a1;
    }
    lds_fence();           // A reloads done before next step overwrites the tile
  }

  // s[node] = h_last . u   (u = W_fc @ W_gcn, in ws[0..63])
  const float* u = ws;
  if (lane < 16) {
    int node = base + lane;
    if (node < N) {
      float s = 0.0f;
#pragma unroll
      for (int k = 0; k < HDIM; ++k)
        s = fmaf((float)h_lds[wv][lane * HDIM + k], u[k], s);
      s_out[node] = s;
    }
  }
}

// ---------------------------------------------------------------------------
// Graph phase (scalar messages after the GCN+FC collapse)
// ---------------------------------------------------------------------------
__global__ void strnn_initdeg(float* __restrict__ deg, int N) {
  int i = blockIdx.x * blockDim.x + threadIdx.x;
  if (i < N) deg[i] = 1.0f;               // self-loop
}

__global__ void strnn_deg(const int* __restrict__ ei, float* __restrict__ deg, int E) {
  int e = blockIdx.x * blockDim.x + threadIdx.x;
  if (e < E) atomicAdd(&deg[ei[2 * e + 1]], 1.0f);
}

__global__ void strnn_norm(const float* __restrict__ s, const float* __restrict__ deg,
                           const float* __restrict__ ws, float* __restrict__ dinv,
                           float* __restrict__ out, int N) {
  int i = blockIdx.x * blockDim.x + threadIdx.x;
  if (i < N) {
    float d  = deg[i];
    float di = rsqrtf(d);
    dinv[i] = di;
    // cbias + self-loop message: s[i] * dinv[i]^2
    out[i] = fmaf(s[i], di * di, ws[64]);
  }
}

__global__ void strnn_edge(const int* __restrict__ ei, const float* __restrict__ s,
                           const float* __restrict__ dinv, float* __restrict__ out, int E) {
  int e = blockIdx.x * blockDim.x + threadIdx.x;
  if (e < E) {
    int src = ei[2 * e];
    int dst = ei[2 * e + 1];
    atomicAdd(&out[dst], dinv[src] * dinv[dst] * s[src]);
  }
}

// ---------------------------------------------------------------------------
extern "C" void kernel_launch(void* const* d_in, const int* in_sizes, int n_in,
                              void* d_out, int out_size, void* d_ws, size_t ws_size,
                              hipStream_t stream) {
  const float* x    = (const float*)d_in[0];
  const int*   ei   = (const int*)  d_in[1];
  const float* Wih  = (const float*)d_in[2];
  const float* bih  = (const float*)d_in[3];
  const float* Whh  = (const float*)d_in[4];
  const float* bhh  = (const float*)d_in[5];
  const float* Wgcn = (const float*)d_in[6];
  const float* bgcn = (const float*)d_in[7];
  const float* Wfc  = (const float*)d_in[8];
  const float* bfc  = (const float*)d_in[9];
  float* out = (float*)d_out;

  const int N = out_size;                 // out is (N,1) f32
  const int T = in_sizes[0] / N;          // x is (N,T,1)
  const int E = in_sizes[1] / 2;          // edge_index is (E,2)

  float* wsf  = (float*)d_ws;             // [0..63]=u, [64]=cbias
  float* s    = wsf + 128;                // N floats
  float* deg  = s + N;                    // N floats
  float* dinv = deg + N;                  // N floats

  const int TPB   = 256;
  const int gN    = (N + TPB - 1) / TPB;
  const int gE    = (E + TPB - 1) / TPB;
  const int gRNN  = (N + 127) / 128;      // 8 waves/block x 16 nodes/wave

  strnn_prep   <<<1,    64,  0, stream>>>(Wgcn, bgcn, Wfc, bfc, wsf);
  strnn_initdeg<<<gN,   TPB, 0, stream>>>(deg, N);
  strnn_rnn    <<<gRNN, TPB, 0, stream>>>(x, Wih, bih, Whh, bhh, wsf, s, N, T);
  strnn_deg    <<<gE,   TPB, 0, stream>>>(ei, deg, E);
  strnn_norm   <<<gN,   TPB, 0, stream>>>(s, deg, wsf, dinv, out, N);
  strnn_edge   <<<gE,   TPB, 0, stream>>>(ei, s, dinv, out, E);
}
